// UpBlock_62947040690362
// MI455X (gfx1250) — matrix operand order repaired and checked by reading, not compiled, not run on hardware
//
#include <hip/hip_runtime.h>

// ---------------------------------------------------------------------------
// UpBlock (MeshUpMP unpool + 2x InteractionNetwork) for MI455X / gfx1250.
//
// All GEMMs run on v_wmma_f32_16x16x32_bf16 (f32 accumulate). Weights are
// repacked on-device into WMMA B-fragment order (ISA 7.12.2 layouts) so the
// inner loop reads one contiguous 32B LDS chunk per lane per fragment.
// Weight staging to LDS uses global_load_async_to_lds_b128 (ASYNCcnt).
// A fragments are gathered directly from the f32 feature rows (fusing the
// v[cluster]/v[s]/v[r] gathers and concats into the GEMM). segment_sum uses
// global f32 atomics. Hidden activations pass between the two MLP layers
// through a per-wave LDS tile, fenced with s_wait_dscnt (CDNA5 split wait).
// ---------------------------------------------------------------------------

#define DEVFN static __device__ __forceinline__

typedef __attribute__((ext_vector_type(16))) __bf16 v16bf;
typedef __attribute__((ext_vector_type(8)))  float  v8f;

static constexpr int kNFine = 65536;
static constexpr int kNEdge = 393216;
static constexpr int kW     = 128;

// fragment sizes (elements)
static constexpr int kFragElems = 512;          // 32 lanes * 16 bf16
// packed sizes per matrix: KT*8 fragments
static constexpr int kFragsU1 = 4 * 8, kFragsU2 = 4 * 8;   // unpool 128->128, 128->128
static constexpr int kFragsE1 = 12 * 8, kFragsE2 = 4 * 8;  // edge 384->128, 128->128
static constexpr int kFragsN1 = 8 * 8, kFragsN2 = 4 * 8;   // node 256->128, 128->128

DEVFN __bf16 f2bf(float f) { return (__bf16)f; }  // native cvt (RNE)

DEVFN float selu_f(float x) {
  // Branchless exact selu: for x>0 the exp term is 0, for x<=0 the max term
  // is 0. Avoids EXEC-mask divergence around v_exp_f32.
  float p = fmaxf(x, 0.f);
  float r = fminf(x, 0.f);
  return 1.0507009873554805f * p + 1.7580993408473766f * (__expf(r) - 1.f);
}

// A fragment (16x32 bf16, ISA 7.12.2): lane L (m = L&15), elems i<8 -> K=k0+i,
// i>=8 -> K=k0+16+i-8, where caller passes k0 = 32*kt + (L>=16 ? 8 : 0).
DEVFN v16bf afrag_f32(const float* rp, int k0) {
  const float4* q0 = (const float4*)(rp + k0);        // 32B aligned
  const float4* q1 = (const float4*)(rp + k0 + 16);
  float4 x0 = q0[0], x1 = q0[1], x2 = q1[0], x3 = q1[1];
  v16bf a;
  a[0] = f2bf(x0.x);  a[1] = f2bf(x0.y);  a[2] = f2bf(x0.z);  a[3] = f2bf(x0.w);
  a[4] = f2bf(x1.x);  a[5] = f2bf(x1.y);  a[6] = f2bf(x1.z);  a[7] = f2bf(x1.w);
  a[8] = f2bf(x2.x);  a[9] = f2bf(x2.y);  a[10] = f2bf(x2.z); a[11] = f2bf(x2.w);
  a[12] = f2bf(x3.x); a[13] = f2bf(x3.y); a[14] = f2bf(x3.z); a[15] = f2bf(x3.w);
  return a;
}

DEVFN v16bf afrag_lds(const __bf16* rp, int k0) {
  v16bf a;
#pragma unroll
  for (int i = 0; i < 8; ++i) a[i] = rp[k0 + i];
#pragma unroll
  for (int i = 0; i < 8; ++i) a[i + 8] = rp[k0 + 16 + i];
  return a;
}

// Cooperative global->LDS copy of packed weights via the CDNA5 async path
// (GLOBAL_LOAD_ASYNC_TO_LDS_B128, tracked by ASYNCcnt). The LDS destination
// VGPR is the wave-relative LDS byte offset = low 32 bits of the generic
// pointer (flat LDS aperture keeps the offset in addr[31:0]).
// bytes % (256*16) need not be 0 (per-thread loop), 256 threads.
DEVFN void copy_w(__bf16* dst, const __bf16* src, int bytes) {
  unsigned lds_base = (unsigned)(uintptr_t)dst;
  const int n = bytes >> 4;  // 16B chunks
  for (int c = threadIdx.x; c < n; c += 256) {
    unsigned dofs = lds_base + (unsigned)(c << 4);
    const char* g = (const char*)src + ((size_t)c << 4);
    asm volatile("global_load_async_to_lds_b128 %0, %1, off"
                 :: "v"(dofs), "v"(g) : "memory");
  }
  asm volatile("s_wait_asynccnt 0" ::: "memory");
  __syncthreads();
}

// Two-layer MLP on a 16-row tile, one wave:
//   hid = selu(A1 @ W1 + b1 [+ relv ⊗ w1x])   (KT1*32 -> 128)
//   out = hid @ W2 + b2                        (128 -> 128)
// sW = [W1 frags | W2 frags] in LDS, hid = per-wave 16x128 bf16 LDS tile.
template <int KT1, bool EXTRA>
DEVFN void mlp2(const v16bf (&a1)[KT1], const __bf16* sW,
                const float* b1, const float* b2,
                __bf16* hid, int lane,
                const float* w1x, const float (&relv)[8],
                v8f (&outacc)[8]) {
  const int nl = lane & 15;
  const int hi = lane >> 4;
#pragma unroll
  for (int nt = 0; nt < 8; ++nt) {
    float bb = b1[nt * 16 + nl];
    v8f acc;
#pragma unroll
    for (int j = 0; j < 8; ++j) acc[j] = bb;
#pragma unroll
    for (int kt = 0; kt < KT1; ++kt) {
      v16bf bw = *(const v16bf*)(sW + ((size_t)(kt * 8 + nt) * 32 + lane) * 16);
      acc = __builtin_amdgcn_wmma_f32_16x16x32_bf16(
          false, a1[kt], false, bw, (short)0, acc, false, false);
    }
    if (EXTRA) {  // rank-1 rel_pos column of the unpool W1
      float wx = w1x[nt * 16 + nl];
#pragma unroll
      for (int j = 0; j < 8; ++j) acc[j] += relv[j] * wx;
    }
    // C layout: elem j -> (m = j + 8*hi, n = nt*16 + nl)
#pragma unroll
    for (int j = 0; j < 8; ++j)
      hid[(j + (hi << 3)) * 128 + nt * 16 + nl] = f2bf(selu_f(acc[j]));
  }
  asm volatile("s_wait_dscnt 0" ::: "memory");  // same-wave LDS RAW fence
  const __bf16* hrd = hid + nl * 128;
  v16bf a2[4];
#pragma unroll
  for (int kt = 0; kt < 4; ++kt) a2[kt] = afrag_lds(hrd, (kt << 5) + (hi << 3));
  const __bf16* sW2 = sW + (size_t)KT1 * 8 * kFragElems;
#pragma unroll
  for (int nt = 0; nt < 8; ++nt) {
    float bb = b2[nt * 16 + nl];
    v8f acc;
#pragma unroll
    for (int j = 0; j < 8; ++j) acc[j] = bb;
#pragma unroll
    for (int kt = 0; kt < 4; ++kt) {
      v16bf bw = *(const v16bf*)(sW2 + ((size_t)(kt * 8 + nt) * 32 + lane) * 16);
      acc = __builtin_amdgcn_wmma_f32_16x16x32_bf16(
          false, a2[kt], false, bw, (short)0, acc, false, false);
    }
    outacc[nt] = acc;
  }
}

// ---------------------------------------------------------------------------
// Pack a K x 128 row-major f32 weight matrix into bf16 B-fragment order.
// B frag (32x16): lane L holds column n = nt*16 + (L&15); elems i<8 -> K =
// kt*32 + (L>=16 ? 8 : 0) + i, i>=8 -> that + 8. One thread per (fragment, lane).
// ---------------------------------------------------------------------------
__global__ __launch_bounds__(256) void pack_b(const float* __restrict__ Wm,
                                              __bf16* __restrict__ out, int KT) {
  int tid = blockIdx.x * 256 + threadIdx.x;
  int total = KT * 8 * 32;
  if (tid >= total) return;
  int f = tid >> 5, L = tid & 31;
  int kt = f >> 3, nt = f & 7;
  int kbase = kt * 32 + ((L >> 4) << 3);
  int n = nt * 16 + (L & 15);
  __bf16* o = out + (size_t)tid * 16;
#pragma unroll
  for (int i = 0; i < 16; ++i) o[i] = f2bf(Wm[(size_t)(kbase + i + (i >= 8 ? 8 : 0)) * 128 + n]);
}

__global__ __launch_bounds__(256) void zero_f32(float* p, int n) {
  for (int i = blockIdx.x * 256 + threadIdx.x; i < n; i += gridDim.x * 256)
    p[i] = 0.f;
}

// ---------------------------------------------------------------------------
// Unpool: vout[row] = MLP([v[cluster[row]], rel[row]]) + c_skip[row]
// ---------------------------------------------------------------------------
__global__ __launch_bounds__(256) void unpool_kernel(
    const float* __restrict__ vc, const int* __restrict__ cluster,
    const float* __restrict__ rel, const float* __restrict__ cskip,
    const __bf16* __restrict__ pW, const float* __restrict__ b1,
    const float* __restrict__ b2, const float* __restrict__ w1full,
    float* __restrict__ vout) {
  extern __shared__ char smem[];
  __bf16* sW = (__bf16*)smem;
  copy_w(sW, pW, (kFragsU1 + kFragsU2) * kFragElems * 2);
  __bf16* sHid = (__bf16*)(smem + (kFragsU1 + kFragsU2) * kFragElems * 2);

  const int lane = threadIdx.x & 31, wave = threadIdx.x >> 5;
  const int m = lane & 15, hi = lane >> 4;
  const size_t tileBase = ((size_t)blockIdx.x * 8 + wave) * 16;

  const float* arow = vc + (size_t)cluster[tileBase + m] * 128;
  v16bf a1[4];
#pragma unroll
  for (int kt = 0; kt < 4; ++kt) a1[kt] = afrag_f32(arow, (kt << 5) + (hi << 3));
  float relv[8];
#pragma unroll
  for (int j = 0; j < 8; ++j) relv[j] = rel[tileBase + j + (hi << 3)];

  v8f outacc[8];
  mlp2<4, true>(a1, sW, b1, b2, sHid + wave * 2048, lane, w1full + 128 * 128,
                relv, outacc);
#pragma unroll
  for (int nt = 0; nt < 8; ++nt) {
    int col = nt * 16 + m;
#pragma unroll
    for (int j = 0; j < 8; ++j) {
      size_t ro = tileBase + j + (hi << 3);
      vout[ro * 128 + col] = outacc[nt][j] + cskip[ro * 128 + col];
    }
  }
}

// ---------------------------------------------------------------------------
// Edge layer: e_out[k] = e_in[k] + MLP([e_in[k], v[s[k]], v[r[k]]]);
// atomically accumulates e_out rows into agg[r[k]] (segment_sum).
// e_in may alias e_out (per-element read-then-write by one lane).
// ---------------------------------------------------------------------------
__global__ __launch_bounds__(256) void edge_kernel(
    const float* e_in, float* e_out, const float* __restrict__ vfeat,
    const int* __restrict__ sidx, const int* __restrict__ ridx,
    const __bf16* __restrict__ pW, const float* __restrict__ b1,
    const float* __restrict__ b2, float* __restrict__ agg) {
  extern __shared__ char smem[];
  __bf16* sW = (__bf16*)smem;
  copy_w(sW, pW, (kFragsE1 + kFragsE2) * kFragElems * 2);
  __bf16* sHid = (__bf16*)(smem + (kFragsE1 + kFragsE2) * kFragElems * 2);

  const int lane = threadIdx.x & 31, wave = threadIdx.x >> 5;
  const int m = lane & 15, hi = lane >> 4;
  const size_t tileBase = ((size_t)blockIdx.x * 8 + wave) * 16;
  const size_t rowA = tileBase + m;

  const float* erow = e_in + rowA * 128;
  const float* vs = vfeat + (size_t)sidx[rowA] * 128;
  const float* vr = vfeat + (size_t)ridx[rowA] * 128;

  v16bf a1[12];  // K = 384 = [e | v_s | v_r]
#pragma unroll
  for (int kt = 0; kt < 12; ++kt) {
    const float* rp = (kt < 4) ? erow : (kt < 8) ? vs : vr;
    a1[kt] = afrag_f32(rp, ((kt & 3) << 5) + (hi << 3));
  }
  int r2[8];
#pragma unroll
  for (int j = 0; j < 8; ++j) r2[j] = ridx[tileBase + j + (hi << 3)];

  float dummy[8] = {};
  v8f outacc[8];
  mlp2<12, false>(a1, sW, b1, b2, sHid + wave * 2048, lane, b1, dummy, outacc);

#pragma unroll
  for (int nt = 0; nt < 8; ++nt) {
    int col = nt * 16 + m;
#pragma unroll
    for (int j = 0; j < 8; ++j) {
      size_t ro = tileBase + j + (hi << 3);
      float x = outacc[nt][j] + e_in[ro * 128 + col];
      e_out[ro * 128 + col] = x;
      unsafeAtomicAdd(&agg[(size_t)r2[j] * 128 + col], x);  // segment_sum
    }
  }
}

// ---------------------------------------------------------------------------
// Node layer: vout[row] = vin[row] + MLP([vin[row], agg[row]])
// vin may alias vout.
// ---------------------------------------------------------------------------
__global__ __launch_bounds__(256) void node_kernel(
    const float* vin, const float* __restrict__ agg,
    const __bf16* __restrict__ pW, const float* __restrict__ b1,
    const float* __restrict__ b2, float* vout) {
  extern __shared__ char smem[];
  __bf16* sW = (__bf16*)smem;
  copy_w(sW, pW, (kFragsN1 + kFragsN2) * kFragElems * 2);
  __bf16* sHid = (__bf16*)(smem + (kFragsN1 + kFragsN2) * kFragElems * 2);

  const int lane = threadIdx.x & 31, wave = threadIdx.x >> 5;
  const int m = lane & 15, hi = lane >> 4;
  const size_t tileBase = ((size_t)blockIdx.x * 8 + wave) * 16;
  const size_t rowA = tileBase + m;

  const float* vrow = vin + rowA * 128;
  const float* arow = agg + rowA * 128;
  v16bf a1[8];  // K = 256 = [v | agg]
#pragma unroll
  for (int kt = 0; kt < 8; ++kt) {
    const float* rp = (kt < 4) ? vrow : arow;
    a1[kt] = afrag_f32(rp, ((kt & 3) << 5) + (hi << 3));
  }
  float dummy[8] = {};
  v8f outacc[8];
  mlp2<8, false>(a1, sW, b1, b2, sHid + wave * 2048, lane, b1, dummy, outacc);

#pragma unroll
  for (int nt = 0; nt < 8; ++nt) {
    int col = nt * 16 + m;
#pragma unroll
    for (int j = 0; j < 8; ++j) {
      size_t ro = tileBase + j + (hi << 3);
      vout[ro * 128 + col] = outacc[nt][j] + vin[ro * 128 + col];
    }
  }
}

// ---------------------------------------------------------------------------
extern "C" void kernel_launch(void* const* d_in, const int* in_sizes, int n_in,
                              void* d_out, int out_size, void* d_ws,
                              size_t ws_size, hipStream_t stream) {
  const float* v      = (const float*)d_in[0];
  const float* c_skip = (const float*)d_in[1];
  const float* e_skip = (const float*)d_in[2];
  const float* rel    = (const float*)d_in[3];
  const float* uw1 = (const float*)d_in[4];
  const float* ub1 = (const float*)d_in[5];
  const float* uw2 = (const float*)d_in[6];
  const float* ub2 = (const float*)d_in[7];
  const float* ew1 = (const float*)d_in[8];
  const float* eb1 = (const float*)d_in[9];
  const float* ew2 = (const float*)d_in[10];
  const float* eb2 = (const float*)d_in[11];
  const float* nw1 = (const float*)d_in[12];
  const float* nb1 = (const float*)d_in[13];
  const float* nw2 = (const float*)d_in[14];
  const float* nb2 = (const float*)d_in[15];
  const int* eidx = (const int*)d_in[16];
  const int* sidx = eidx;
  const int* ridx = eidx + kNEdge;
  const int* cluster = (const int*)d_in[17];
  // d_in[18] = batch_skip (unused, single graph)

  // workspace layout (bytes); ~257 MB total
  char* ws = (char*)d_ws;
  __bf16* pU  = (__bf16*)(ws + 0);          // 65536  B packed unpool [W1|W2]
  __bf16* pE0 = (__bf16*)(ws + 65536);      // 131072 B packed edge L0
  __bf16* pE1 = (__bf16*)(ws + 196608);     // 131072 B packed edge L1
  __bf16* pN0 = (__bf16*)(ws + 327680);     // 98304  B packed node L0
  __bf16* pN1 = (__bf16*)(ws + 425984);     // 98304  B packed node L1
  float* agg   = (float*)(ws + 524288);     // 32 MB
  float* vwork = (float*)(ws + 524288 + 33554432ull);
  float* ebuf  = (float*)(ws + 524288 + 67108864ull);

  // --- pack weights into WMMA fragment order (bf16) ---
  pack_b<<<4, 256, 0, stream>>>(uw1, pU, 4);  // rows 0..127 of [129,128]
  pack_b<<<4, 256, 0, stream>>>(uw2, pU + kFragsU1 * kFragElems, 4);
  const __bf16* pE[2] = {pE0, pE1};
  const __bf16* pN[2] = {pN0, pN1};
  for (int l = 0; l < 2; ++l) {
    pack_b<<<12, 256, 0, stream>>>(ew1 + (size_t)l * 384 * 128, (__bf16*)pE[l], 12);
    pack_b<<<4, 256, 0, stream>>>(ew2 + (size_t)l * 128 * 128,
                                  (__bf16*)pE[l] + kFragsE1 * kFragElems, 4);
    pack_b<<<8, 256, 0, stream>>>(nw1 + (size_t)l * 256 * 128, (__bf16*)pN[l], 8);
    pack_b<<<4, 256, 0, stream>>>(nw2 + (size_t)l * 128 * 128,
                                  (__bf16*)pN[l] + kFragsN1 * kFragElems, 4);
  }

  // --- unpool MLP + c_skip ---
  unpool_kernel<<<kNFine / 128, 256, 98304, stream>>>(
      v, cluster, rel, c_skip, pU, ub1, ub2, uw1, vwork);

  // --- 2x InteractionNetwork ---
  for (int l = 0; l < 2; ++l) {
    zero_f32<<<4096, 256, 0, stream>>>(agg, kNFine * kW);
    edge_kernel<<<kNEdge / 128, 256, 163840, stream>>>(
        l == 0 ? e_skip : ebuf, ebuf, vwork, sidx, ridx, pE[l],
        eb1 + (size_t)l * 128, eb2 + (size_t)l * 128, agg);
    node_kernel<<<kNFine / 128, 256, 131072, stream>>>(
        vwork, agg, pN[l], nb1 + (size_t)l * 128, nb2 + (size_t)l * 128,
        l == 0 ? vwork : (float*)d_out);
  }
}
